// SparseMOE_18476949307432
// MI455X (gfx1250) — matrix operand order repaired and checked by reading, not compiled, not run on hardware
//
#include <hip/hip_runtime.h>

// ---------------- problem constants ----------------
#define BSZ    4
#define SEQ    2048
#define TOK    (BSZ * SEQ)          // 8192 tokens
#define DIN    1024
#define DOUT   1024
#define NEXP   8
#define MT64   (TOK / 64)           // 128 worst-case 64-token tiles per expert

typedef __attribute__((ext_vector_type(16))) __bf16 v16bf;
typedef __attribute__((ext_vector_type(8)))  float  v8f;

union Frag16 { v16bf v; uint4 q[2]; };

static __device__ __forceinline__ unsigned short f2bf(float f) {
    union { float f; unsigned u; } c; c.f = f;
    unsigned u = c.u;
    return (unsigned short)((u + 0x7FFFu + ((u >> 16) & 1u)) >> 16); // RNE
}

// ---------------- x: fp32 -> bf16 ----------------
__global__ void moe_cvt_x(const float* __restrict__ x, unsigned short* __restrict__ xb) {
    size_t i = ((size_t)blockIdx.x * blockDim.x + threadIdx.x) * 8;
    const float4* p = (const float4*)(x + i);
    float4 a = p[0], b = p[1];
    union { unsigned short s[8]; uint4 q; } o;
    o.s[0] = f2bf(a.x); o.s[1] = f2bf(a.y); o.s[2] = f2bf(a.z); o.s[3] = f2bf(a.w);
    o.s[4] = f2bf(b.x); o.s[5] = f2bf(b.y); o.s[6] = f2bf(b.z); o.s[7] = f2bf(b.w);
    *(uint4*)(xb + i) = o.q;
}

// ---------------- w1/w3: fp32 [E][K][N] -> bf16 [E][N][K] (tiled transpose) ----------------
__global__ void moe_cvt_w(const float* __restrict__ w1, const float* __restrict__ w3,
                          unsigned short* __restrict__ w1t, unsigned short* __restrict__ w3t) {
    __shared__ float tile[32][33];
    int z = blockIdx.z, e = z >> 1, which = z & 1;
    const float*     src = (which ? w3 : w1)  + (size_t)e * DIN * DOUT;
    unsigned short*  dst = (which ? w3t : w1t) + (size_t)e * DIN * DOUT;
    int k0 = blockIdx.y << 5, n0 = blockIdx.x << 5;
    int t = threadIdx.x, tr = t >> 5, tc = t & 31;
#pragma unroll
    for (int j = 0; j < 4; ++j) {
        int k = tr + j * 8;
        tile[k][tc] = src[(size_t)(k0 + k) * DOUT + n0 + tc];
    }
    __syncthreads();
#pragma unroll
    for (int j = 0; j < 4; ++j) {
        int n = tr + j * 8;
        dst[(size_t)(n0 + n) * DIN + k0 + tc] = f2bf(tile[tc][n]);
    }
}

// ---------------- gating: logits, top-2, softmax, scatter to expert lists ----------------
__global__ void moe_gate(const float* __restrict__ x, const float* __restrict__ wg,
                         int* __restrict__ counts, int* __restrict__ tok_ids,
                         float* __restrict__ gatebuf) {
    int wv = threadIdx.x >> 5, lane = threadIdx.x & 31;
    int t = blockIdx.x * 8 + wv;
    const float* xr = x + (size_t)t * DIN;
    float acc[NEXP] = {};
#pragma unroll 4
    for (int i = 0; i < 32; ++i) {
        int k = i * 32 + lane;
        float xv = xr[k];
        const float4* wr = (const float4*)(wg + (size_t)k * NEXP);
        float4 w0 = wr[0], w1v = wr[1];
        acc[0] += xv * w0.x;  acc[1] += xv * w0.y;  acc[2] += xv * w0.z;  acc[3] += xv * w0.w;
        acc[4] += xv * w1v.x; acc[5] += xv * w1v.y; acc[6] += xv * w1v.z; acc[7] += xv * w1v.w;
    }
#pragma unroll
    for (int off = 16; off > 0; off >>= 1)
#pragma unroll
        for (int e = 0; e < NEXP; ++e) acc[e] += __shfl_xor(acc[e], off, 32);
    if (lane == 0) {
        float best = -1e30f; int bi = 0;
#pragma unroll
        for (int e = 0; e < NEXP; ++e) if (acc[e] > best) { best = acc[e]; bi = e; }
        float sec = -1e30f; int si = (bi == 0) ? 1 : 0;
#pragma unroll
        for (int e = 0; e < NEXP; ++e) if (e != bi && acc[e] > sec) { sec = acc[e]; si = e; }
        float gb = 1.0f / (1.0f + __expf(sec - best));   // softmax over top-2
        float gs = 1.0f - gb;
        int p0 = atomicAdd(&counts[bi], 1);
        tok_ids[bi * TOK + p0] = t;  gatebuf[bi * TOK + p0] = gb;
        int p1 = atomicAdd(&counts[si], 1);
        tok_ids[si * TOK + p1] = t;  gatebuf[si * TOK + p1] = gs;
    }
}

// ---------------- grouped expert GEMM ----------------
// Block: 256 threads (8 waves). Tile: M=64 gathered tokens x N=128 (16/wave), K=1024.
// A (tokens) streamed to double-buffered LDS via global_load_async_to_lds_b128 (ASYNCcnt).
// B (weights, bf16 [E][N][K]) register-prefetched one k-step ahead; each B fragment pair
// feeds 8 WMMAs (4 M-subtiles x {w1,w3}).
#define KCH   128
#define LDSW  (KCH + 8)             // row stride 272B: 16B-aligned, conflict-free b128
#define ABUFB (64 * LDSW * 2)       // one LDS A-buffer in bytes

__global__ __launch_bounds__(256) void moe_gemm(
        const unsigned short* __restrict__ xb,
        const unsigned short* __restrict__ w1t,
        const unsigned short* __restrict__ w3t,
        const int* __restrict__ counts,
        const int* __restrict__ tok_ids,
        const float* __restrict__ gatebuf,
        float* __restrict__ out) {
    int e    = blockIdx.x >> 7;            // / MT64
    int tile = blockIdx.x & (MT64 - 1);
    int cnt  = counts[e];
    if (tile * 64 >= cnt) return;

    __shared__ unsigned short Atile[2][64][LDSW];
    __shared__ float s_gate[64];
    __shared__ int   s_tok[64];

    int t = threadIdx.x;
    if (t < 64) {
        int idx = tile * 64 + t;
        if (idx < cnt) { s_tok[t] = tok_ids[e * TOK + idx]; s_gate[t] = gatebuf[e * TOK + idx]; }
        else           { s_tok[t] = tok_ids[e * TOK + tile * 64]; s_gate[t] = 0.0f; }
    }
    __syncthreads();

    // ---- per-thread async-staging addresses: 4 rows (srow+16j), 8 bf16 each ----
    int srow = t >> 4, sseg = t & 15;
    unsigned           lofs[4];
    unsigned long long gptr[4];
#pragma unroll
    for (int j = 0; j < 4; ++j) {
        int row = srow + j * 16;
        lofs[j] = (unsigned)(size_t)&Atile[0][row][sseg * 8];
        gptr[j] = (unsigned long long)(xb + (size_t)s_tok[row] * DIN + sseg * 8);
    }

#define ISSUE_CHUNK(CH, BUF)                                                        \
    {                                                                               \
        _Pragma("unroll")                                                           \
        for (int j = 0; j < 4; ++j) {                                               \
            unsigned la = lofs[j] + ((BUF) ? ABUFB : 0);                            \
            unsigned long long ga = gptr[j] + (unsigned long long)(CH) * (KCH * 2); \
            asm volatile("global_load_async_to_lds_b128 %0, %1, off"                \
                         :: "v"(la), "v"(ga) : "memory");                           \
        }                                                                           \
    }

    int wv = t >> 5, lane = t & 31;
    int m  = lane & 15, hi = lane >> 4;
    int n  = (blockIdx.y << 7) + wv * 16 + m;        // this lane's output column

    // rolling B pointers: fragment = 2 x uint4 at [n*K + step*32 + hi*16]
    const uint4* w1p = (const uint4*)w1t + (((size_t)e * DOUT + n) * DIN >> 3) + hi * 2;
    const uint4* w3p = (const uint4*)w3t + (((size_t)e * DOUT + n) * DIN >> 3) + hi * 2;

    v8f c1[4] = {v8f{}, v8f{}, v8f{}, v8f{}};
    v8f c3[4] = {v8f{}, v8f{}, v8f{}, v8f{}};

    Frag16 b1c, b3c, b1n, b3n;
    b1c.q[0] = w1p[0]; b1c.q[1] = w1p[1];
    b3c.q[0] = w3p[0]; b3c.q[1] = w3p[1];

    ISSUE_CHUNK(0, 0);                                // prime chunk 0 -> buf 0

    for (int ch = 0; ch < DIN / KCH; ++ch) {
        if (ch < DIN / KCH - 1) {
            ISSUE_CHUNK(ch + 1, (ch + 1) & 1);        // overlap next chunk with compute
            asm volatile("s_wait_asynccnt 0x4" ::: "memory");  // older 4 (chunk ch) done
        } else {
            asm volatile("s_wait_asynccnt 0x0" ::: "memory");
        }
        __syncthreads();                              // chunk ch visible block-wide

        const unsigned short (*buf)[LDSW] = Atile[ch & 1];
#pragma unroll
        for (int ks = 0; ks < KCH / 32; ++ks) {
            int gstep = ch * (KCH / 32) + ks;
            int ni = gstep + 1; if (ni > 31) ni = 31;     // clamp tail prefetch
            b1n.q[0] = w1p[ni * 4]; b1n.q[1] = w1p[ni * 4 + 1];
            b3n.q[0] = w3p[ni * 4]; b3n.q[1] = w3p[ni * 4 + 1];

            Frag16 a[4];
            int base = ks * 32 + hi * 16;                 // contiguous k order: k = 16*hi + s
#pragma unroll
            for (int mt = 0; mt < 4; ++mt) {
                a[mt].q[0] = *(const uint4*)(&buf[mt * 16 + m][base]);
                a[mt].q[1] = *(const uint4*)(&buf[mt * 16 + m][base + 8]);
            }
#pragma unroll
            for (int mt = 0; mt < 4; ++mt) {
                c1[mt] = __builtin_amdgcn_wmma_f32_16x16x32_bf16(
                            false, a[mt].v, false, b1c.v, (short)0, c1[mt], false, false);
                c3[mt] = __builtin_amdgcn_wmma_f32_16x16x32_bf16(
                            false, a[mt].v, false, b3c.v, (short)0, c3[mt], false, false);
            }
            b1c = b1n; b3c = b3n;
        }
        __syncthreads();                              // all waves done with buf before refill
    }

    // epilogue: gate * silu(h1) * h3, scatter-add (each out element gets exactly 2 adds)
#pragma unroll
    for (int mt = 0; mt < 4; ++mt) {
#pragma unroll
        for (int j = 0; j < 8; ++j) {
            int mrow = mt * 16 + j + hi * 8;          // C/D layout: VGPR j -> M = j + 8*hi
            float g = s_gate[mrow];
            if (g != 0.0f) {
                float h1 = c1[mt][j], h3 = c3[mt][j];
                float v = g * (h1 / (1.0f + __expf(-h1))) * h3;
                atomicAdd(&out[(size_t)s_tok[mrow] * DOUT + n], v);
            }
        }
    }
#undef ISSUE_CHUNK
}

// ---------------- host side ----------------
extern "C" void kernel_launch(void* const* d_in, const int* in_sizes, int n_in,
                              void* d_out, int out_size, void* d_ws, size_t ws_size,
                              hipStream_t stream) {
    const float* x  = (const float*)d_in[0];
    const float* wg = (const float*)d_in[1];
    const float* w1 = (const float*)d_in[2];
    const float* w3 = (const float*)d_in[3];
    float* out = (float*)d_out;

    char* ws = (char*)d_ws;
    int*            counts  = (int*)ws;                                   // 32 B
    int*            tok_ids = (int*)(ws + 256);                           // 256 KB
    float*          gatebuf = (float*)(ws + 256 + NEXP * TOK * 4);        // 256 KB
    unsigned short* xb  = (unsigned short*)(ws + 524544);                 // 16 MB
    unsigned short* w1t = xb  + (size_t)TOK * DIN;                        // 16 MB
    unsigned short* w3t = w1t + (size_t)NEXP * DIN * DOUT;                // 16 MB

    hipMemsetAsync(counts, 0, NEXP * sizeof(int), stream);
    hipMemsetAsync(d_out, 0, (size_t)out_size * sizeof(float), stream);

    moe_cvt_x<<<(TOK * DIN) / (256 * 8), 256, 0, stream>>>(x, xb);
    moe_cvt_w<<<dim3(DOUT / 32, DIN / 32, NEXP * 2), 256, 0, stream>>>(w1, w3, w1t, w3t);
    moe_gate<<<TOK / 8, 256, 0, stream>>>(x, wg, counts, tok_ids, gatebuf);
    moe_gemm<<<dim3(NEXP * MT64, DOUT / 128), 256, 0, stream>>>(
        xb, w1t, w3t, counts, tok_ids, gatebuf, out);
}
